// Performer_block_52785148068422
// MI455X (gfx1250) — compile-verified, matrix operand order checked by reading, not run on hardware
//
#include <hip/hip_runtime.h>
#include <hip/hip_bf16.h>

// ---------------- problem constants ----------------
#define B_   4
#define N_   2048
#define D_   1024
#define L_   4
#define M_   256
#define H_   16
#define DH_  64
#define Z_   (B_*H_)        // 64 (b,h) pairs
#define CH_  16             // z slices per chunk (== H_, one batch index)
#define NCHUNK_ (Z_/CH_)    // 4
static const long SZ_ = (long)B_*N_*D_;   // 8388608

typedef _Float16 h16;
typedef __attribute__((ext_vector_type(16))) _Float16 v16h;
typedef __attribute__((ext_vector_type(8)))  _Float16 v8h;
typedef __attribute__((ext_vector_type(8)))  float    v8f;

// ---------------- WMMA fragment load (unguarded: all tiles exact) ----------------
// A operand layout (16-bit, 16x32): lane l holds row M=l%16.
// lanes 0-15: K=0..7 (f[0..7]), K=16..23 (f[8..15])
// lanes 16-31: K=8..15, K=24..31  -> per-lane base offset 8*(l>>4), +16 for hi half.
// B operand uses the symmetric layout indexed by output column (row-major (col,K) array).
__device__ inline v16h load16(const h16* p) {
  v16h f;
  v8h lo = *(const v8h*)p;
  v8h hh = *(const v8h*)(p + 16);
#pragma unroll
  for (int i = 0; i < 8; i++) { f[i] = lo[i]; f[i + 8] = hh[i]; }
  return f;
}

// ---------------- generic batched WMMA GEMM ----------------
// C[z][row,col] = sum_k A[z][row,k] * B[z][col,k]  (+bias[col]) (+gelu) (+res)
// REQUIREMENT (met by every launch below): R % (WR*64) == 0 within the grid,
// Cc % (WC*64) == 0 within the grid, K % 32 == 0 -> no bounds guards anywhere.
struct GemmArgs {
  const h16* A;  long lda; long aSz;
  const h16* Bm; long ldb; long bSz;
  float* C; h16* C16; long ldc; long cSz;
  const float* bias;
  const float* res;
  int K;
  int gelu;
};

template<int WR, int WC>
__global__ void __launch_bounds__(WR*WC*32)
gemm_wmma(GemmArgs g) {
  const int lane = threadIdx.x & 31;
  const int w    = threadIdx.x >> 5;
  const int wr   = w / WC, wc = w % WC;
  const long z   = blockIdx.z;
  const int rowW = blockIdx.y * (WR * 64) + wr * 64;
  const int colW = blockIdx.x * (WC * 64) + wc * 64;
  const int lo16 = lane & 15, hi = lane >> 4;

  // per-fragment row pointers (K-loop just adds k0)
  const h16* aptr[4];
  const h16* bptr[4];
  {
    const h16* A  = g.A  + z * g.aSz;
    const h16* Bm = g.Bm + z * g.bSz;
#pragma unroll
    for (int i = 0; i < 4; i++)
      aptr[i] = A + (long)(rowW + 16 * i + lo16) * g.lda + 8 * hi;
#pragma unroll
    for (int j = 0; j < 4; j++)
      bptr[j] = Bm + (long)(colW + 16 * j + lo16) * g.ldb + 8 * hi;
  }

  v8f acc[4][4];
#pragma unroll
  for (int i = 0; i < 4; i++)
#pragma unroll
    for (int j = 0; j < 4; j++)
#pragma unroll
      for (int e = 0; e < 8; e++) acc[i][j][e] = 0.f;

  for (int k0 = 0; k0 < g.K; k0 += 32) {
    v16h a[4], b[4];
#pragma unroll
    for (int i = 0; i < 4; i++) a[i] = load16(aptr[i] + k0);
#pragma unroll
    for (int j = 0; j < 4; j++) b[j] = load16(bptr[j] + k0);
#pragma unroll
    for (int i = 0; i < 4; i++)
#pragma unroll
      for (int j = 0; j < 4; j++)
        acc[i][j] = __builtin_amdgcn_wmma_f32_16x16x32_f16(
            false, a[i], false, b[j], (short)0, acc[i][j], false, false);
  }

  float* C   = g.C   ? g.C   + z * g.cSz : nullptr;
  h16*  C16  = g.C16 ? g.C16 + z * g.cSz : nullptr;
  const float* res = g.res ? g.res + z * g.cSz : nullptr;
#pragma unroll
  for (int i = 0; i < 4; i++) {
#pragma unroll
    for (int j = 0; j < 4; j++) {
      int col = colW + 16 * j + lo16;
      float bv = g.bias ? g.bias[col] : 0.f;
#pragma unroll
      for (int r = 0; r < 8; r++) {
        int row = rowW + 16 * i + 8 * hi + r;
        long idx = (long)row * g.ldc + col;
        float v = acc[i][j][r] + bv;
        if (g.gelu) v = 0.5f * v * (1.f + erff(v * 0.70710678118f));
        if (res) v += res[idx];
        if (C)   C[idx]   = v;
        if (C16) C16[idx] = (h16)v;
      }
    }
  }
}

// ---------------- elementwise / reduction kernels ----------------
__global__ void __launch_bounds__(256) copy_f32(const float* s, float* d, long n) {
  long i = (long)blockIdx.x * 256 + threadIdx.x;
  if (i < n) d[i] = s[i];
}
__global__ void __launch_bounds__(256) cvt_f16(const float* s, h16* d, long n) {
  long i = (long)blockIdx.x * 256 + threadIdx.x;
  if (i < n) d[i] = (h16)s[i];
}

// LayerNorm over D=1024, write f16 (one block of 256 per row)
__global__ void __launch_bounds__(256) ln_f16(const float* x, const float* g,
                                              const float* b, h16* out) {
  long row = blockIdx.x;
  const float* xr = x + row * D_;
  float vals[4], s = 0.f, s2 = 0.f;
#pragma unroll
  for (int i = 0; i < 4; i++) {
    float v = xr[threadIdx.x + 256 * i];
    vals[i] = v; s += v; s2 += v * v;
  }
  __shared__ float sA[256], sB[256];
  sA[threadIdx.x] = s; sB[threadIdx.x] = s2; __syncthreads();
  for (int o = 128; o > 0; o >>= 1) {
    if (threadIdx.x < o) { sA[threadIdx.x] += sA[threadIdx.x+o]; sB[threadIdx.x] += sB[threadIdx.x+o]; }
    __syncthreads();
  }
  float mean = sA[0] * (1.f / D_);
  float var  = sB[0] * (1.f / D_) - mean * mean;
  float rs   = rsqrtf(var + 1e-5f);
#pragma unroll
  for (int i = 0; i < 4; i++) {
    int id = threadIdx.x + 256 * i;
    out[row * D_ + id] = (h16)((vals[i] - mean) * rs * g[id] + b[id]);
  }
}

// scale by dh^-0.25, write f16, and diag = 0.5*sum(xn^2) per (b,h,n)
__global__ void __launch_bounds__(64) qk_prep(const float* q, h16* qh, float* diag) {
  long blk = blockIdx.x;             // over B*N*H
  int  h   = (int)(blk % H_);
  long bn  = blk / H_;
  float v = q[bn * D_ + h * DH_ + threadIdx.x] * 0.35355339059f; // 64^-0.25
  qh[bn * D_ + h * DH_ + threadIdx.x] = (h16)v;
  __shared__ float sm[64];
  sm[threadIdx.x] = v * v; __syncthreads();
  for (int o = 32; o > 0; o >>= 1) {
    if (threadIdx.x < o) sm[threadIdx.x] += sm[threadIdx.x + o];
    __syncthreads();
  }
  if (threadIdx.x == 0) {
    long b = bn / N_, n = bn % N_;
    diag[(b * H_ + h) * (long)N_ + n] = 0.5f * sm[0];
  }
}

// v (b,n,h,d) f32 -> vt (z,d,n) f16, LDS-tiled transpose
__global__ void __launch_bounds__(256) v_transpose(const float* v, h16* vt) {
  __shared__ h16 tile[32][33];
  int z = blockIdx.z, b = z >> 4, h = z & 15;
  int n0 = blockIdx.x * 32, d0 = blockIdx.y * 32;
#pragma unroll
  for (int i = 0; i < 32; i += 8) {
    int n = n0 + threadIdx.y + i;
    tile[threadIdx.y + i][threadIdx.x] =
        (h16)v[((long)(b * N_ + n)) * D_ + h * DH_ + d0 + threadIdx.x];
  }
  __syncthreads();
#pragma unroll
  for (int i = 0; i < 32; i += 8) {
    int d = d0 + threadIdx.y + i;
    vt[((long)z * DH_ + d) * N_ + n0 + threadIdx.x] = tile[threadIdx.x][threadIdx.y + i];
  }
}

// stage-1 global max over (n,m) per z-slice of the dd chunk
__global__ void __launch_bounds__(256) kmax_partial(const float* dd, float* part) {
  int zc = blockIdx.y;
  long base = (long)zc * N_ * M_ + (long)blockIdx.x * 8192;
  float m = -1e30f;
  for (int i = threadIdx.x; i < 8192; i += 256) m = fmaxf(m, dd[base + i]);
  __shared__ float sm[256];
  sm[threadIdx.x] = m; __syncthreads();
  for (int o = 128; o > 0; o >>= 1) {
    if (threadIdx.x < o) sm[threadIdx.x] = fmaxf(sm[threadIdx.x], sm[threadIdx.x + o]);
    __syncthreads();
  }
  if (threadIdx.x == 0) part[zc * 64 + blockIdx.x] = sm[0];
}

// keys: kpT[z,m,n] = f16( M^-0.5 * (exp(dd - diag - stab) + eps) )
__global__ void __launch_bounds__(256) kexp(const float* dd, const float* part,
                                            const float* diagk, h16* kpT, int zBase) {
  int zc = blockIdx.y, n = blockIdx.x, z = zBase + zc;
  __shared__ float s_stab;
  if (threadIdx.x == 0) {
    float m = -1e30f;
    for (int i = 0; i < 64; i++) m = fmaxf(m, part[zc * 64 + i]);
    s_stab = m;
  }
  __syncthreads();
  float d = dd[((long)zc * N_ + n) * M_ + threadIdx.x];
  float val = 0.0625f * (expf(d - diagk[(long)z * N_ + n] - s_stab) + 1e-4f);
  kpT[((long)z * M_ + threadIdx.x) * N_ + n] = (h16)val;
}

// queries: row-max stab, qp f16, and d_inv = 1/dot(qp,ksum)
__global__ void __launch_bounds__(256) qexp(const float* dd, const float* diagq,
                                            const float* ksum, h16* qp, float* dinv,
                                            int zBase) {
  int zc = blockIdx.y, n = blockIdx.x, z = zBase + zc;
  float d = dd[((long)zc * N_ + n) * M_ + threadIdx.x];
  __shared__ float sm[256];
  sm[threadIdx.x] = d; __syncthreads();
  for (int o = 128; o > 0; o >>= 1) {
    if (threadIdx.x < o) sm[threadIdx.x] = fmaxf(sm[threadIdx.x], sm[threadIdx.x + o]);
    __syncthreads();
  }
  float stab = sm[0]; __syncthreads();
  float val = 0.0625f * (expf(d - diagq[(long)z * N_ + n] - stab) + 1e-4f);
  qp[((long)z * N_ + n) * M_ + threadIdx.x] = (h16)val;
  sm[threadIdx.x] = val * ksum[(long)z * M_ + threadIdx.x]; __syncthreads();
  for (int o = 128; o > 0; o >>= 1) {
    if (threadIdx.x < o) sm[threadIdx.x] += sm[threadIdx.x + o];
    __syncthreads();
  }
  if (threadIdx.x == 0) dinv[(long)z * N_ + n] = 1.f / sm[0];
}

// ksum[z,m] = sum_n kpT[z,m,n]
__global__ void __launch_bounds__(256) ksum_kernel(const h16* kpT, float* ksum) {
  long row = blockIdx.x;            // z*M + m
  const h16* p = kpT + row * N_;
  float s = 0.f;
  for (int i = threadIdx.x; i < N_; i += 256) s += (float)p[i];
  __shared__ float sm[256];
  sm[threadIdx.x] = s; __syncthreads();
  for (int o = 128; o > 0; o >>= 1) {
    if (threadIdx.x < o) sm[threadIdx.x] += sm[threadIdx.x + o];
    __syncthreads();
  }
  if (threadIdx.x == 0) ksum[row] = sm[0];
}

// ctx (z,m,d) f32 -> ctxT (z,d,m) f16
__global__ void __launch_bounds__(256) ctx_transpose(const float* ctx, h16* ctxT) {
  long i = (long)blockIdx.x * 256 + threadIdx.x;
  long tot = (long)Z_ * M_ * DH_;
  if (i >= tot) return;
  long z = i / (M_ * DH_), r = i % (M_ * DH_), m = r / DH_, d = r % DH_;
  ctxT[(z * DH_ + d) * M_ + m] = (h16)ctx[i];
}

// out2 (z,n,d)*dinv -> attn (b,n,h*64+d) f16
__global__ void __launch_bounds__(256) attn_merge(const float* out2, const float* dinv,
                                                  h16* attn) {
  long i = (long)blockIdx.x * 256 + threadIdx.x;
  if (i >= SZ_) return;
  long z = i / ((long)N_ * DH_), r = i % ((long)N_ * DH_), n = r / DH_, d = r % DH_;
  long b = z >> 4, h = z & 15;
  attn[(b * N_ + n) * (long)D_ + h * DH_ + d] = (h16)(out2[i] * dinv[z * N_ + n]);
}

// ---------------- host orchestration ----------------
static inline long cdiv(long a, long b) { return (a + b - 1) / b; }

extern "C" void kernel_launch(void* const* d_in, const int* in_sizes, int n_in,
                              void* d_out, int out_size, void* d_ws, size_t ws_size,
                              hipStream_t stream) {
  const float* x_in  = (const float*)d_in[0];
  const float* projW = (const float*)d_in[1];
  const float* ln1g  = (const float*)d_in[2];
  const float* ln1b  = (const float*)d_in[3];
  const float* Wq    = (const float*)d_in[4];
  const float* bq    = (const float*)d_in[5];
  const float* Wk    = (const float*)d_in[6];
  const float* bk    = (const float*)d_in[7];
  const float* Wv    = (const float*)d_in[8];
  const float* bv    = (const float*)d_in[9];
  const float* Wo    = (const float*)d_in[10];
  const float* bo    = (const float*)d_in[11];
  const float* ln2g  = (const float*)d_in[12];
  const float* ln2b  = (const float*)d_in[13];
  const float* W1    = (const float*)d_in[14];
  const float* b1    = (const float*)d_in[15];
  const float* W2    = (const float*)d_in[16];
  const float* b2    = (const float*)d_in[17];

  char* p = (char*)d_ws;
  auto alloc = [&](size_t bytes) -> char* {
    char* r = p; p += (bytes + 255) & ~(size_t)255; return r;
  };
  float* xcur   = (float*)alloc(SZ_ * 4);
  h16*   hf16   = (h16*)  alloc(SZ_ * 2);
  h16*   h1f16  = (h16*)  alloc(SZ_ * 2);
  float* qbuf   = (float*)alloc(SZ_ * 4);
  float* kbuf   = (float*)alloc(SZ_ * 4);
  float* vbuf   = (float*)alloc(SZ_ * 4);
  h16*   qf16   = (h16*)  alloc(SZ_ * 2);
  h16*   kf16   = (h16*)  alloc(SZ_ * 2);
  h16*   vtf16  = (h16*)  alloc(SZ_ * 2);
  float* ddbuf  = (float*)alloc((long)CH_ * N_ * M_ * 4);
  h16*   qp16   = (h16*)  alloc((long)Z_ * N_ * M_ * 2);
  h16*   kpT16  = (h16*)  alloc((long)Z_ * M_ * N_ * 2);
  float* ctx    = (float*)alloc((long)Z_ * M_ * DH_ * 4);
  h16*   ctxT16 = (h16*)  alloc((long)Z_ * DH_ * M_ * 2);
  float* out2   = (float*)alloc((long)Z_ * N_ * DH_ * 4);
  h16*   attn16 = (h16*)  alloc(SZ_ * 2);
  float* diagq  = (float*)alloc((long)Z_ * N_ * 4);
  float* diagk  = (float*)alloc((long)Z_ * N_ * 4);
  float* dinv   = (float*)alloc((long)Z_ * N_ * 4);
  float* ksum   = (float*)alloc((long)Z_ * M_ * 4);
  float* kmaxp  = (float*)alloc((long)CH_ * 64 * 4);
  h16*   wtmp   = (h16*)  alloc((long)D_ * D_ * 2);
  h16*   proj16 = (h16*)  alloc((long)M_ * DH_ * 2);
  if ((size_t)(p - (char*)d_ws) > ws_size) return;  // insufficient workspace

  const long BN = (long)B_ * N_;

  copy_f32<<<dim3(cdiv(SZ_, 256)), 256, 0, stream>>>(x_in, xcur, SZ_);

  // main (BN x D) x (D x D) GEMM: 64x4 blocks of 128x256, exact tiling
  auto gemmMain = [&](const h16* A, const h16* W, const float* bias,
                      float* C, h16* C16, const float* res, int gelu) {
    GemmArgs g;
    g.A = A;  g.lda = D_; g.aSz = 0;
    g.Bm = W; g.ldb = D_; g.bSz = 0;
    g.C = C; g.C16 = C16; g.ldc = D_; g.cSz = 0;
    g.bias = bias; g.res = res;
    g.K = D_; g.gelu = gelu;
    gemm_wmma<2, 4><<<dim3(D_ / 256, BN / 128, 1), 256, 0, stream>>>(g);
  };

  for (int l = 0; l < L_; l++) {
    // ---- attention sublayer ----
    ln_f16<<<dim3(BN), 256, 0, stream>>>(xcur, ln1g + (long)l * D_, ln1b + (long)l * D_, hf16);

    cvt_f16<<<dim3(cdiv((long)D_ * D_, 256)), 256, 0, stream>>>(Wq + (long)l * D_ * D_, wtmp, (long)D_ * D_);
    gemmMain(hf16, wtmp, bq + (long)l * D_, qbuf, nullptr, nullptr, 0);
    cvt_f16<<<dim3(cdiv((long)D_ * D_, 256)), 256, 0, stream>>>(Wk + (long)l * D_ * D_, wtmp, (long)D_ * D_);
    gemmMain(hf16, wtmp, bk + (long)l * D_, kbuf, nullptr, nullptr, 0);
    cvt_f16<<<dim3(cdiv((long)D_ * D_, 256)), 256, 0, stream>>>(Wv + (long)l * D_ * D_, wtmp, (long)D_ * D_);
    gemmMain(hf16, wtmp, bv + (long)l * D_, vbuf, nullptr, nullptr, 0);

    cvt_f16<<<dim3(cdiv((long)M_ * DH_, 256)), 256, 0, stream>>>(projW + (long)l * M_ * DH_, proj16, (long)M_ * DH_);
    qk_prep<<<dim3(BN * H_), 64, 0, stream>>>(qbuf, qf16, diagq);
    qk_prep<<<dim3(BN * H_), 64, 0, stream>>>(kbuf, kf16, diagk);
    v_transpose<<<dim3(N_ / 32, DH_ / 32, Z_), dim3(32, 8), 0, stream>>>(vbuf, vtf16);

    // feature-map projection GEMM (per chunk of CH_ z slices): 2048x256, K=64
    auto projGemm = [&](const h16* src, int c) {
      GemmArgs g;
      g.A = src + (long)c * N_ * D_; g.lda = D_; g.aSz = DH_;
      g.Bm = proj16; g.ldb = DH_; g.bSz = 0;
      g.C = ddbuf; g.C16 = nullptr; g.ldc = M_; g.cSz = (long)N_ * M_;
      g.bias = nullptr; g.res = nullptr;
      g.K = DH_; g.gelu = 0;
      gemm_wmma<2, 4><<<dim3(1, N_ / 128, CH_), 256, 0, stream>>>(g);   // 256 cols exact
    };

    // ---- key path: dd -> global-max -> kpT f16 ----
    for (int c = 0; c < NCHUNK_; c++) {
      projGemm(kf16, c);
      kmax_partial<<<dim3(64, CH_), 256, 0, stream>>>(ddbuf, kmaxp);
      kexp<<<dim3(N_, CH_), 256, 0, stream>>>(ddbuf, kmaxp, diagk, kpT16, c * CH_);
    }
    ksum_kernel<<<dim3((long)Z_ * M_), 256, 0, stream>>>(kpT16, ksum);

    // ctx[z] = kpT (256 x 2048) * vT (2048 x 64): 256 rows / 64 cols exact
    {
      GemmArgs g;
      g.A = kpT16; g.lda = N_; g.aSz = (long)M_ * N_;
      g.Bm = vtf16; g.ldb = N_; g.bSz = (long)DH_ * N_;
      g.C = ctx; g.C16 = nullptr; g.ldc = DH_; g.cSz = (long)M_ * DH_;
      g.bias = nullptr; g.res = nullptr;
      g.K = N_; g.gelu = 0;
      gemm_wmma<4, 1><<<dim3(1, 1, Z_), 128, 0, stream>>>(g);
    }
    ctx_transpose<<<dim3(cdiv((long)Z_ * M_ * DH_, 256)), 256, 0, stream>>>(ctx, ctxT16);

    // ---- query path: dd -> row-max -> qp f16 + d_inv ----
    for (int c = 0; c < NCHUNK_; c++) {
      projGemm(qf16, c);
      qexp<<<dim3(N_, CH_), 256, 0, stream>>>(ddbuf, diagq, ksum, qp16, dinv, c * CH_);
    }

    // out2[z] = qp (2048 x 256) * ctxT' (256 x 64): exact tiling
    {
      GemmArgs g;
      g.A = qp16; g.lda = M_; g.aSz = (long)N_ * M_;
      g.Bm = ctxT16; g.ldb = M_; g.bSz = (long)DH_ * M_;
      g.C = out2; g.C16 = nullptr; g.ldc = DH_; g.cSz = (long)N_ * DH_;
      g.bias = nullptr; g.res = nullptr;
      g.K = M_; g.gelu = 0;
      gemm_wmma<4, 1><<<dim3(1, N_ / 256, Z_), 128, 0, stream>>>(g);
    }
    attn_merge<<<dim3(cdiv(SZ_, 256)), 256, 0, stream>>>(out2, dinv, attn16);

    // output projection + residual (in place on xcur)
    cvt_f16<<<dim3(cdiv((long)D_ * D_, 256)), 256, 0, stream>>>(Wo + (long)l * D_ * D_, wtmp, (long)D_ * D_);
    gemmMain(attn16, wtmp, bo + (long)l * D_, xcur, nullptr, xcur, 0);

    // ---- FFN sublayer ----
    ln_f16<<<dim3(BN), 256, 0, stream>>>(xcur, ln2g + (long)l * D_, ln2b + (long)l * D_, hf16);
    cvt_f16<<<dim3(cdiv((long)D_ * D_, 256)), 256, 0, stream>>>(W1 + (long)l * D_ * D_, wtmp, (long)D_ * D_);
    gemmMain(hf16, wtmp, b1 + (long)l * D_, nullptr, h1f16, nullptr, 1);   // fused exact GELU
    cvt_f16<<<dim3(cdiv((long)D_ * D_, 256)), 256, 0, stream>>>(W2 + (long)l * D_ * D_, wtmp, (long)D_ * D_);
    gemmMain(h1f16, wtmp, b2 + (long)l * D_, xcur, nullptr, xcur, 0);
  }

  copy_f32<<<dim3(cdiv(SZ_, 256)), 256, 0, stream>>>(xcur, (float*)d_out, SZ_);
}
